// inpainting_44933947850872
// MI455X (gfx1250) — compile-verified, matrix operand order checked by reading, not run on hardware
//
#include <hip/hip_runtime.h>
#include <hip/hip_bf16.h>

// ---------------------------------------------------------------------------
// Problem constants (from reference shapes)
// ---------------------------------------------------------------------------
#define BDIM   128      // batch (b) and also o-range of softmax'd region
#define CDIM   512      // channel dim (contraction K of the GEMMs)
#define LM     256      // mid last dim
#define LL     256      // left last dim
#define LS     512      // shortcut last dim
#define TDIM   128      // nit (number of t GEMMs)
#define OUT_L  512      // output last dim
#define KC     64       // K-chunk staged in LDS per iteration
#define NCH    (CDIM / KC)

typedef __attribute__((ext_vector_type(16))) __bf16 v16bf;
typedef __attribute__((ext_vector_type(8)))  float  v8f;

static __device__ __forceinline__ unsigned short f2bf(float f) {
    unsigned int u = __float_as_uint(f);
    u += 0x7fffu + ((u >> 16) & 1u);          // round-to-nearest-even
    return (unsigned short)(u >> 16);
}

struct U4 { unsigned int x, y, z, w; };
union FragCast { U4 u[2]; v16bf v; };

// Load a 16x32 bf16 WMMA operand fragment from a row-major [128][KC] LDS tile.
// Lane L: row = rowbase + (L&15); half h = L>>4 selects k-subsets
//   VGPRs 0-3: k = kb + 8h .. +7 ; VGPRs 4-7: k = kb + 16 + 8h .. +7
static __device__ __forceinline__ v16bf load_frag(const unsigned short* tile,
                                                  int row, int kb, int lane) {
    const int h = (lane >> 4) & 1;
    const unsigned short* p = tile + row * KC + kb + 8 * h;
    FragCast fc;
    fc.u[0] = *(const U4*)(p);
    fc.u[1] = *(const U4*)(p + 16);
    return fc.v;
}

static __device__ __forceinline__ v8f wmma_bf16(v16bf a, v16bf b, v8f c) {
    // (neg_a, A, neg_b, B, c_mod, C, reuse_a, reuse_b)
    return __builtin_amdgcn_wmma_f32_16x16x32_bf16(false, a, false, b,
                                                   (short)0, c, false, false);
}

// ---------------------------------------------------------------------------
// Kernel 1: zero the j in [384, 512) tail of both outputs (disjoint region).
// 2 * 128 * 512 rows, 128 floats each -> 32 float4 stores per row.
// ---------------------------------------------------------------------------
__global__ __launch_bounds__(256) void zero_tail_kernel(float* __restrict__ out) {
    size_t i   = (size_t)blockIdx.x * 256 + threadIdx.x;   // [0, 2*128*512*32)
    size_t row = i >> 5;                                   // (raw,b,c) flat row
    size_t q   = i & 31;
    float4 z;  z.x = 0.f; z.y = 0.f; z.z = 0.f; z.w = 0.f;
    *(float4*)(out + row * OUT_L + 384 + q * 4) = z;
}

// ---------------------------------------------------------------------------
// Kernel 2: per-t batched GEMM (3 matmuls sharing operands) + softmax + scatter.
// One workgroup (8 wave32) per t.
// ---------------------------------------------------------------------------
__global__ __launch_bounds__(256) void gemm_softmax_kernel(
        const float* __restrict__ left, const float* __restrict__ mid,
        const float* __restrict__ sc00, const float* __restrict__ sc10,
        float* __restrict__ out) {
    __shared__ unsigned short X0s[BDIM * KC];
    __shared__ unsigned short X1s[BDIM * KC];
    __shared__ unsigned short W0s[BDIM * KC];
    __shared__ unsigned short W1s[BDIM * KC];

    const int t    = blockIdx.x;        // 0..127
    const int tid  = threadIdx.x;       // 0..255
    const int lane = tid & 31;
    const int wave = tid >> 5;          // 0..7
    const int b0   = wave * 16;         // b-strip owned by this wave
    const size_t OUTHALF = (size_t)BDIM * CDIM * OUT_L;

    v8f acc0[8] = {};                   // out0 tiles, o-tile 0..7
    v8f acc1[8] = {};                   // out1 tiles

    for (int ch = 0; ch < NCH; ++ch) {
        // ---- stage K-chunk into LDS (f32 -> bf16) --------------------------
        #pragma unroll 4
        for (int i = 0; i < (BDIM * KC) / 256; ++i) {      // 32 iterations
            int idx = i * 256 + tid;
            int r   = idx >> 6;                            // row 0..127
            int cc  = idx & (KC - 1);
            int c   = ch * KC + cc;
            float2 xm = *(const float2*)(mid  + ((size_t)r * CDIM + c) * LM + 2 * t);
            float2 wl = *(const float2*)(left + ((size_t)r * CDIM + c) * LL + 2 * t);
            X0s[r * KC + cc] = f2bf(xm.x);                 // mid even  -> X0
            X1s[r * KC + cc] = f2bf(xm.y);                 // mid odd   -> X1
            W0s[r * KC + cc] = f2bf(wl.x);                 // left even -> W0
            W1s[r * KC + cc] = f2bf(wl.y);                 // left odd  -> W1
        }
        __syncthreads();

        // prefetch a slice of the next chunk (emits global_prefetch_b8)
        if (ch + 1 < NCH) {
            int idx = tid * 32;
            int r = idx >> 6, cc = idx & (KC - 1);
            int c = (ch + 1) * KC + cc;
            __builtin_prefetch(mid  + ((size_t)r * CDIM + c) * LM + 2 * t, 0, 0);
            __builtin_prefetch(left + ((size_t)r * CDIM + c) * LL + 2 * t, 0, 0);
        }

        // ---- WMMA over this chunk: 2 k-steps x 8 o-tiles x 3 matmuls -------
        #pragma unroll
        for (int kb = 0; kb < KC; kb += 32) {
            v16bf a0 = load_frag(X0s, b0 + (lane & 15), kb, lane);
            v16bf a1 = load_frag(X1s, b0 + (lane & 15), kb, lane);
            #pragma unroll
            for (int ot = 0; ot < 8; ++ot) {
                v16bf bw0 = load_frag(W0s, ot * 16 + (lane & 15), kb, lane);
                v16bf bw1 = load_frag(W1s, ot * 16 + (lane & 15), kb, lane);
                acc0[ot] = wmma_bf16(a0, bw1, acc0[ot]);   // out0 = X0*W1^T
                acc1[ot] = wmma_bf16(a0, bw0, acc1[ot]);   // out1 = X0*W0^T
                acc1[ot] = wmma_bf16(a1, bw1, acc1[ot]);   //       + X1*W1^T
            }
        }
        __syncthreads();
    }

    // ---- register softmax over o (8 tiles x 16 lanes per row) --------------
    // C/D layout: element r of acc[ot], lane L -> b = b0 + r + 8*(L>>4),
    //                                            o = ot*16 + (L&15)
    const int hi = lane >> 4;
    const int n  = lane & 15;
    #pragma unroll
    for (int r = 0; r < 8; ++r) {
        float m0 = -3.0e38f, m1 = -3.0e38f;
        #pragma unroll
        for (int ot = 0; ot < 8; ++ot) {
            m0 = fmaxf(m0, acc0[ot][r]);
            m1 = fmaxf(m1, acc1[ot][r]);
        }
        #pragma unroll
        for (int sh = 1; sh < 16; sh <<= 1) {
            m0 = fmaxf(m0, __shfl_xor(m0, sh, 16));
            m1 = fmaxf(m1, __shfl_xor(m1, sh, 16));
        }
        float e0[8], e1[8], s0 = 0.f, s1 = 0.f;
        #pragma unroll
        for (int ot = 0; ot < 8; ++ot) {
            e0[ot] = __expf(10.f * (acc0[ot][r] - m0));  s0 += e0[ot];
            e1[ot] = __expf(10.f * (acc1[ot][r] - m1));  s1 += e1[ot];
        }
        #pragma unroll
        for (int sh = 1; sh < 16; sh <<= 1) {
            s0 += __shfl_xor(s0, sh, 16);
            s1 += __shfl_xor(s1, sh, 16);
        }
        const float r0 = 1.f / s0, r1 = 1.f / s1;
        const int b = b0 + r + 8 * hi;
        #pragma unroll
        for (int ot = 0; ot < 8; ++ot) {
            const int c = ot * 16 + n;
            const size_t base = ((size_t)b * CDIM + c) * OUT_L + 3 * t;
            const float y0 = e0[ot] * r0, y1 = e1[ot] * r1;
            out[base]               = y0;  out[base + 1]           = y1;
            out[OUTHALF + base]     = y0;  out[OUTHALF + base + 1] = y1;
        }
    }

    // ---- center term + zero yi-slots for c >= 128 (padded softmax region) --
    for (int idx = tid; idx < BDIM * CDIM; idx += 256) {
        const int b = idx >> 9;
        const int c = idx & (CDIM - 1);
        const size_t ib = (size_t)b * CDIM + c;
        const float v0 = sc00[ib * LS + t];     // raw0 center
        const float v1 = sc10[ib * LS + t];     // raw1 center
        const size_t ob = ib * OUT_L + 3 * t;
        out[ob + 2]           = v0;
        out[OUTHALF + ob + 2] = v1;
        if (c >= BDIM) {                        // softmax pad rows are zero
            out[ob]               = 0.f;  out[ob + 1]           = 0.f;
            out[OUTHALF + ob]     = 0.f;  out[OUTHALF + ob + 1] = 0.f;
        }
    }
}

// ---------------------------------------------------------------------------
extern "C" void kernel_launch(void* const* d_in, const int* in_sizes, int n_in,
                              void* d_out, int out_size, void* d_ws, size_t ws_size,
                              hipStream_t stream) {
    // setup_inputs order: left, right, mid, sc00, sc01, sc10, sc11
    const float* left = (const float*)d_in[0];
    const float* mid  = (const float*)d_in[2];
    const float* sc00 = (const float*)d_in[3];
    const float* sc10 = (const float*)d_in[5];
    float* out = (float*)d_out;

    // zero the j >= 384 tail of both outputs: 2*128*512*32 float4 stores
    zero_tail_kernel<<<(2 * BDIM * CDIM * 32) / 256, 256, 0, stream>>>(out);
    // one workgroup per t
    gemm_softmax_kernel<<<TDIM, 256, 0, stream>>>(left, mid, sc00, sc10, out);
}
